// MultiHeadSelfAttention_31782757990445
// MI455X (gfx1250) — matrix operand order repaired and checked by reading, not compiled, not run on hardware
//
#include <hip/hip_runtime.h>
#include <hip/hip_bf16.h>

// ---------------------------------------------------------------------------
// MI455X (gfx1250) multi-head self-attention, f16 WMMA path, f32 accumulate.
// B=8, N=1024, D=768, H=12, Hd=64.
// - Operands pre-converted/pre-transposed once (everything lives in 192MB L2).
// - QKV GEMM + attention: LDS-free WMMA with contiguous v8h fragment loads.
// - Projection GEMM: TDM (tensor_load_to_lds) double-buffered LDS pipeline.
// ---------------------------------------------------------------------------

typedef __attribute__((ext_vector_type(16))) _Float16 v16h;
typedef __attribute__((ext_vector_type(8)))  _Float16 v8h;
typedef __attribute__((ext_vector_type(8)))  float    v8f;
typedef __attribute__((ext_vector_type(4)))  unsigned int u32x4;
typedef __attribute__((ext_vector_type(8)))  int          i32x8;
typedef __attribute__((ext_vector_type(4)))  int          i32x4;

union H16 { v16h v; v8h h[2]; };

static __device__ __forceinline__ v8f wmma_f32_f16(v16h a, v16h b, v8f c) {
  // D = A(16x32 f16) * B(32x16 f16) + C(16x16 f32)
  return __builtin_amdgcn_wmma_f32_16x16x32_f16(
      /*neg_a=*/false, a, /*neg_b=*/false, b,
      /*c_mod=*/(short)0, c, /*reuse_a=*/false, /*reuse_b=*/false);
}

#define BATCH 8
#define SEQ   1024
#define DIM   768
#define HEADS 12
#define HDIM  64
#define DQKV  2304
#define ROWS  (BATCH * SEQ)        // 8192
#define LDP   40                   // LDS row stride in halves (80B, 16B aligned)

// ---------------------------------------------------------------------------
// Tensor Data Mover support (gfx1250). 5-arg builtin on ROCm7.2/clang-22,
// 6-arg on clang-23 toolchains. Falls back to direct-global loads if absent.
// ---------------------------------------------------------------------------
#if defined(__has_builtin)
#if __has_builtin(__builtin_amdgcn_tensor_load_to_lds) && \
    __has_builtin(__builtin_amdgcn_s_wait_tensorcnt)
#define HAVE_TDM 1
#endif
#endif

#ifdef HAVE_TDM
// Build D# per CDNA5 ISA sec.8 and issue a 2D tile load:
//   copies tile_d1 rows x tile_d0 f16 elements (row stride = stride_d0 elems)
//   from global tile start `gptr` into LDS at byte offset `lds_off`, packed
//   contiguously (row pitch in LDS = tile_d0 * 2 bytes).
static __device__ __forceinline__ void tdm_load_tile_f16(
    unsigned lds_off, const void* gptr,
    unsigned tensor_d0, unsigned tensor_d1,
    unsigned tile_d0, unsigned tile_d1, unsigned stride_d0) {
  unsigned long long ga = (unsigned long long)gptr;
  u32x4 g0;
  g0[0] = 1u;                                          // count=1 (valid D#)
  g0[1] = lds_off;                                     // lds_addr (bytes)
  g0[2] = (unsigned)(ga & 0xFFFFFFFFu);                // global_addr[31:0]
  g0[3] = (unsigned)((ga >> 32) & 0x01FFFFFFu) | (2u << 30);  // [56:32]|type=2
  i32x8 g1;
  g1[0] = (int)(1u << 16);                             // data_size=1 -> 2 bytes
  g1[1] = (int)((tensor_d0 & 0xFFFFu) << 16);          // tensor_dim0[15:0]
  g1[2] = (int)(((tensor_d0 >> 16) & 0xFFFFu) | ((tensor_d1 & 0xFFFFu) << 16));
  g1[3] = (int)(((tensor_d1 >> 16) & 0xFFFFu) | ((tile_d0 & 0xFFFFu) << 16));
  g1[4] = (int)(tile_d1 & 0xFFFFu);                    // tile_dim1, tile_dim2=0
  g1[5] = (int)stride_d0;                              // dim0_stride[31:0]
  g1[6] = 0;                                           // dim0_stride[47:32]
  g1[7] = 0;                                           // dim1_stride (2D tile)
  i32x4 z4 = {0, 0, 0, 0};
#if __clang_major__ >= 23
  i32x8 z8 = {0, 0, 0, 0, 0, 0, 0, 0};
  __builtin_amdgcn_tensor_load_to_lds(g0, g1, z4, z4, z8, 0);
#else
  __builtin_amdgcn_tensor_load_to_lds(g0, g1, z4, z4, 0);
#endif
}
#endif  // HAVE_TDM

// ---------------------------------------------------------------------------
// Kernel 0a: vectorized f32 -> f16 convert (8 elements / thread)
// ---------------------------------------------------------------------------
__global__ void __launch_bounds__(256) cvt_vec_kernel(
    const float* __restrict__ src, _Float16* __restrict__ dst, int n8) {
  int i = blockIdx.x * 256 + threadIdx.x;
  if (i < n8) {
    const float4* s = (const float4*)src;
    float4 a = s[2 * i], b = s[2 * i + 1];
    v8h o = {(_Float16)a.x, (_Float16)a.y, (_Float16)a.z, (_Float16)a.w,
             (_Float16)b.x, (_Float16)b.y, (_Float16)b.z, (_Float16)b.w};
    ((v8h*)dst)[i] = o;
  }
}

// ---------------------------------------------------------------------------
// Kernel 0b: tiled transpose + convert: src[R][C] f32 -> dst[C][R] f16
// ---------------------------------------------------------------------------
__global__ void __launch_bounds__(256) cvtT_kernel(
    const float* __restrict__ src, _Float16* __restrict__ dst, int R, int C) {
  __shared__ float tile[32][33];
  const int c0 = blockIdx.x * 32, r0 = blockIdx.y * 32;
  const int tx = threadIdx.x & 31, ty = threadIdx.x >> 5;  // 32 x 8
  #pragma unroll
  for (int i = 0; i < 32; i += 8)
    tile[ty + i][tx] = src[(size_t)(r0 + ty + i) * C + c0 + tx];
  __syncthreads();
  #pragma unroll
  for (int i = 0; i < 32; i += 8)
    dst[(size_t)(c0 + ty + i) * R + r0 + tx] = (_Float16)tile[tx][ty + i];
}

// ---------------------------------------------------------------------------
// Kernel 1: QKV GEMM  qkv[8192,2304] = xH[8192,768] @ wT^T + bias
//   LDS-free: A = xH[m][k], B = wT[n][k]; fragments = contiguous v8h pairs.
//   Wave tile 64x64 (4x4 WMMA tiles), block = 4 waves = 128x128 tile.
// Scatters:  Q,K -> [B,H,N,64] f16 ;  V -> [B,H,64,N] f16 (transposed).
// ---------------------------------------------------------------------------
__global__ void __launch_bounds__(128) qkv_gemm_kernel(
    const _Float16* __restrict__ xh,    // [8192,768] f16
    const _Float16* __restrict__ wt,    // [2304,768] f16 (wT[n][k])
    const float*    __restrict__ bias,  // [2304] f32
    _Float16* __restrict__ qh, _Float16* __restrict__ kh,
    _Float16* __restrict__ vth) {
  const int lane = threadIdx.x & 31, wave = threadIdx.x >> 5;
  const int lr = lane & 15, hi = lane >> 4, off = hi * 8;
  const int gm = blockIdx.y * 128 + (wave >> 1) * 64;
  const int gn = blockIdx.x * 128 + (wave & 1) * 64;

  const _Float16* arow[4];
  const _Float16* brow[4];
  #pragma unroll
  for (int i = 0; i < 4; ++i) arow[i] = xh + (size_t)(gm + i * 16 + lr) * DIM;
  #pragma unroll
  for (int j = 0; j < 4; ++j) brow[j] = wt + (size_t)(gn + j * 16 + lr) * DIM;

  v8f acc[4][4] = {};

  for (int k0 = 0; k0 < DIM; k0 += 32) {
    H16 a[4], b[4];
    #pragma unroll
    for (int i = 0; i < 4; ++i) {
      a[i].h[0] = *(const v8h*)(arow[i] + k0 + off);
      a[i].h[1] = *(const v8h*)(arow[i] + k0 + 16 + off);
    }
    #pragma unroll
    for (int j = 0; j < 4; ++j) {
      b[j].h[0] = *(const v8h*)(brow[j] + k0 + off);
      b[j].h[1] = *(const v8h*)(brow[j] + k0 + 16 + off);
    }
    #pragma unroll
    for (int i = 0; i < 4; ++i)
      #pragma unroll
      for (int j = 0; j < 4; ++j)
        acc[i][j] = wmma_f32_f16(a[i].v, b[j].v, acc[i][j]);
  }

  // Epilogue: bias + scatter to Q / K / V^T (f16)
  #pragma unroll
  for (int j = 0; j < 4; ++j) {
    const int col = gn + j * 16 + lr;  // 0..2303
    const float bv = bias[col];
    const int sec = col / DIM;         // 0=Q 1=K 2=V
    const int rem = col % DIM;
    const int h = rem >> 6, d = rem & 63;
    #pragma unroll
    for (int i = 0; i < 4; ++i) {
      #pragma unroll
      for (int r = 0; r < 8; ++r) {
        const int row = gm + i * 16 + hi * 8 + r;  // 0..8191
        const int bb = row >> 10, n = row & 1023;
        const int bhn = bb * HEADS + h;
        const _Float16 hv = (_Float16)(acc[i][j][r] + bv);
        if (sec == 0)
          qh[((size_t)bhn * SEQ + n) * HDIM + d] = hv;
        else if (sec == 1)
          kh[((size_t)bhn * SEQ + n) * HDIM + d] = hv;
        else
          vth[((size_t)bhn * HDIM + d) * SEQ + n] = hv;
      }
    }
  }
}

// ---------------------------------------------------------------------------
// Kernel 2: flash attention. Block = 4 waves, each wave owns 16 query rows.
// Loop over 32-key chunks: 4 score WMMAs + online softmax + 4 PV WMMAs.
// ---------------------------------------------------------------------------
__global__ void __launch_bounds__(128) attention_kernel(
    const _Float16* __restrict__ qh,   // [B*H, N, 64]
    const _Float16* __restrict__ kh,   // [B*H, N, 64]
    const _Float16* __restrict__ vth,  // [B*H, 64, N]
    _Float16* __restrict__ attn) {     // [B, N, 768]
  __shared__ alignas(16) _Float16 Ps[4][16 * LDP];  // per-wave P tile 16x32
  const int lane = threadIdx.x & 31, wave = threadIdx.x >> 5;
  const int lr = lane & 15, hi = lane >> 4, off = hi * 8;
  const int bh = blockIdx.y;                 // 0..95
  const int b = bh / HEADS, h = bh % HEADS;
  const int qBase = blockIdx.x * 64 + wave * 16;
  const _Float16* qp = qh + (size_t)bh * SEQ * HDIM;
  const _Float16* kp = kh + (size_t)bh * SEQ * HDIM;
  const _Float16* vp = vth + (size_t)bh * HDIM * SEQ;

  // Q A-fragments, loaded once (Hd=64 -> two 16x32 fragments)
  H16 qa[2];
  {
    const _Float16* qrow = qp + (size_t)(qBase + lr) * HDIM;
    #pragma unroll
    for (int kk = 0; kk < 2; ++kk) {
      qa[kk].h[0] = *(const v8h*)(qrow + kk * 32 + off);
      qa[kk].h[1] = *(const v8h*)(qrow + kk * 32 + 16 + off);
    }
  }

  float mrow[8], lrow[8];
  v8f o[4] = {};
  #pragma unroll
  for (int r = 0; r < 8; ++r) { mrow[r] = -1e30f; lrow[r] = 0.0f; }

  _Float16* ps = &Ps[wave][0];

  for (int mc = 0; mc < SEQ / 32; ++mc) {
    const int keyBase = mc * 32;
    v8f s0 = {}, s1 = {};
    // Scores: S[16q x 32k] = Q @ K^T, two key tiles, two k-depth steps each
    #pragma unroll
    for (int kk = 0; kk < 2; ++kk) {
      H16 b0, b1;
      const _Float16* kr0 = kp + (size_t)(keyBase + lr) * HDIM + kk * 32 + off;
      const _Float16* kr1 = kp + (size_t)(keyBase + 16 + lr) * HDIM + kk * 32 + off;
      b0.h[0] = *(const v8h*)(kr0); b0.h[1] = *(const v8h*)(kr0 + 16);
      b1.h[0] = *(const v8h*)(kr1); b1.h[1] = *(const v8h*)(kr1 + 16);
      s0 = wmma_f32_f16(qa[kk].v, b0.v, s0);
      s1 = wmma_f32_f16(qa[kk].v, b1.v, s1);
    }
    // Prefetch next K chunk (global_prefetch_b8)
    if (mc + 1 < SEQ / 32)
      __builtin_prefetch(kp + (size_t)(keyBase + 32 + lr) * HDIM, 0, 1);

    // Row max over this chunk (C layout: lane = column, VGPR = row)
    const float scale = 0.125f;  // 1/sqrt(64)
    float cmax[8];
    #pragma unroll
    for (int r = 0; r < 8; ++r) {
      s0[r] *= scale; s1[r] *= scale;
      cmax[r] = fmaxf(s0[r], s1[r]);
    }
    #pragma unroll
    for (int d = 1; d < 16; d <<= 1)
      #pragma unroll
      for (int r = 0; r < 8; ++r)
        cmax[r] = fmaxf(cmax[r], __shfl_xor(cmax[r], d, 32));

    // Online softmax update
    float alpha[8], rsum[8];
    #pragma unroll
    for (int r = 0; r < 8; ++r) {
      const float mn = fmaxf(mrow[r], cmax[r]);
      alpha[r] = __expf(mrow[r] - mn);
      mrow[r] = mn;
      const float p0 = __expf(s0[r] - mn);
      const float p1 = __expf(s1[r] - mn);
      s0[r] = p0; s1[r] = p1;
      rsum[r] = p0 + p1;
    }
    #pragma unroll
    for (int d = 1; d < 16; d <<= 1)
      #pragma unroll
      for (int r = 0; r < 8; ++r)
        rsum[r] += __shfl_xor(rsum[r], d, 32);
    #pragma unroll
    for (int r = 0; r < 8; ++r)
      lrow[r] = lrow[r] * alpha[r] + rsum[r];
    #pragma unroll
    for (int t = 0; t < 4; ++t)
      #pragma unroll
      for (int r = 0; r < 8; ++r)
        o[t][r] *= alpha[r];

    // C-layout -> A-layout transpose of P through per-wave LDS (in-order DS)
    #pragma unroll
    for (int r = 0; r < 8; ++r) {
      const int row = hi * 8 + r;
      ps[row * LDP + lr]      = (_Float16)s0[r];
      ps[row * LDP + 16 + lr] = (_Float16)s1[r];
    }
    H16 pa;
    {
      const _Float16* pr = ps + lr * LDP;
      pa.h[0] = *(const v8h*)(pr + off);
      pa.h[1] = *(const v8h*)(pr + 16 + off);
    }

    // O += P @ V : V^T layout gives contiguous per-lane B fragments
    #pragma unroll
    for (int t = 0; t < 4; ++t) {
      H16 bv;
      const _Float16* vr = vp + (size_t)(t * 16 + lr) * SEQ + keyBase + off;
      bv.h[0] = *(const v8h*)(vr); bv.h[1] = *(const v8h*)(vr + 16);
      o[t] = wmma_f32_f16(pa.v, bv.v, o[t]);
    }
  }

  // Epilogue: normalize, store f16 to [B, N, 768] for the projection GEMM
  #pragma unroll
  for (int t = 0; t < 4; ++t) {
    #pragma unroll
    for (int r = 0; r < 8; ++r) {
      const int row = qBase + hi * 8 + r;
      const float v = o[t][r] / lrow[r];
      attn[((size_t)b * SEQ + row) * DIM + h * HDIM + t * 16 + lr] = (_Float16)v;
    }
  }
}

// ---------------------------------------------------------------------------
// Kernel 3: output projection  out[8192,768] = attn @ proj_w + bias (f32 out)
// TDM path: block tile 128x128, double-buffered LDS slices (128x32 f16 each)
// fetched by tensor_load_to_lds (wave 0 issues, TENSORcnt-tracked), WMMA
// consumes from LDS. Fallback: direct-global fragments (same math).
// ---------------------------------------------------------------------------
__global__ void __launch_bounds__(128) proj_gemm_kernel(
    const _Float16* __restrict__ a,     // [8192,768] f16
    const _Float16* __restrict__ wt,    // [768,768] f16 (wT[n][k])
    const float*    __restrict__ bias,  // [768]
    float* __restrict__ out) {          // [8192,768] f32
  const int lane = threadIdx.x & 31, wave = threadIdx.x >> 5;
  const int lr = lane & 15, hi = lane >> 4, off = hi * 8;

#ifdef HAVE_TDM
  __shared__ alignas(16) _Float16 Abuf[2][128 * 32];  // [m][k] slice, 8KB each
  __shared__ alignas(16) _Float16 Bbuf[2][128 * 32];  // [n][k] slice, 8KB each
  const int gm = blockIdx.y * 128;
  const int gn = blockIdx.x * 128;
  const int mw = (wave >> 1) * 64, nw = (wave & 1) * 64;

  v8f acc[4][4] = {};

  // Preload k-slice 0 into buffer 0 (two 128x32 f16 tile DMAs).
  if (wave == 0) {
    tdm_load_tile_f16((unsigned)(size_t)&Abuf[0][0], a + (size_t)gm * DIM,
                      DIM, ROWS, 32, 128, DIM);
    tdm_load_tile_f16((unsigned)(size_t)&Bbuf[0][0], wt + (size_t)gn * DIM,
                      DIM, DIM, 32, 128, DIM);
    __builtin_amdgcn_s_wait_tensorcnt(0);
  }
  __syncthreads();

  for (int ks = 0; ks < DIM / 32; ++ks) {
    const int buf = ks & 1;
    // Kick off next k-slice into the other buffer while computing this one.
    if (wave == 0 && ks + 1 < DIM / 32) {
      const int k1 = (ks + 1) * 32;
      tdm_load_tile_f16((unsigned)(size_t)&Abuf[buf ^ 1][0],
                        a + (size_t)gm * DIM + k1, DIM, ROWS, 32, 128, DIM);
      tdm_load_tile_f16((unsigned)(size_t)&Bbuf[buf ^ 1][0],
                        wt + (size_t)gn * DIM + k1, DIM, DIM, 32, 128, DIM);
    }

    const _Float16* As = &Abuf[buf][0];
    const _Float16* Bs = &Bbuf[buf][0];
    H16 af[4], bf[4];
    #pragma unroll
    for (int i = 0; i < 4; ++i) {
      const _Float16* ap = As + (mw + i * 16 + lr) * 32;
      af[i].h[0] = *(const v8h*)(ap + off);
      af[i].h[1] = *(const v8h*)(ap + 16 + off);
    }
    #pragma unroll
    for (int j = 0; j < 4; ++j) {
      const _Float16* bp = Bs + (nw + j * 16 + lr) * 32 + off;
      bf[j].h[0] = *(const v8h*)(bp);
      bf[j].h[1] = *(const v8h*)(bp + 16);
    }
    #pragma unroll
    for (int i = 0; i < 4; ++i)
      #pragma unroll
      for (int j = 0; j < 4; ++j)
        acc[i][j] = wmma_f32_f16(af[i].v, bf[j].v, acc[i][j]);

    if (wave == 0 && ks + 1 < DIM / 32)
      __builtin_amdgcn_s_wait_tensorcnt(0);
    __syncthreads();
  }

  #pragma unroll
  for (int j = 0; j < 4; ++j) {
    const int col = gn + nw + j * 16 + lr;
    const float bv = bias[col];
    #pragma unroll
    for (int i = 0; i < 4; ++i) {
      #pragma unroll
      for (int r = 0; r < 8; ++r) {
        const int row = gm + mw + i * 16 + hi * 8 + r;
        out[(size_t)row * DIM + col] = acc[i][j][r] + bv;
      }
    }
  }

#else  // !HAVE_TDM: direct-global fragment loads
  const int gm = blockIdx.y * 128 + (wave >> 1) * 64;
  const int gn = blockIdx.x * 128 + (wave & 1) * 64;

  const _Float16* arow[4];
  const _Float16* brow[4];
  #pragma unroll
  for (int i = 0; i < 4; ++i) arow[i] = a + (size_t)(gm + i * 16 + lr) * DIM;
  #pragma unroll
  for (int j = 0; j < 4; ++j) brow[j] = wt + (size_t)(gn + j * 16 + lr) * DIM;

  v8f acc[4][4] = {};

  for (int k0 = 0; k0 < DIM; k0 += 32) {
    H16 af[4], bf[4];
    #pragma unroll
    for (int i = 0; i < 4; ++i) {
      af[i].h[0] = *(const v8h*)(arow[i] + k0 + off);
      af[i].h[1] = *(const v8h*)(arow[i] + k0 + 16 + off);
    }
    #pragma unroll
    for (int j = 0; j < 4; ++j) {
      bf[j].h[0] = *(const v8h*)(brow[j] + k0 + off);
      bf[j].h[1] = *(const v8h*)(brow[j] + k0 + 16 + off);
    }
    #pragma unroll
    for (int i = 0; i < 4; ++i)
      #pragma unroll
      for (int j = 0; j < 4; ++j)
        acc[i][j] = wmma_f32_f16(af[i].v, bf[j].v, acc[i][j]);
  }

  #pragma unroll
  for (int j = 0; j < 4; ++j) {
    const int col = gn + j * 16 + lr;
    const float bv = bias[col];
    #pragma unroll
    for (int i = 0; i < 4; ++i) {
      #pragma unroll
      for (int r = 0; r < 8; ++r) {
        const int row = gm + i * 16 + hi * 8 + r;
        out[(size_t)row * DIM + col] = acc[i][j][r] + bv;
      }
    }
  }
#endif
}

// ---------------------------------------------------------------------------
// Host launcher
// ---------------------------------------------------------------------------
extern "C" void kernel_launch(void* const* d_in, const int* in_sizes, int n_in,
                              void* d_out, int out_size, void* d_ws, size_t ws_size,
                              hipStream_t stream) {
  const float* x      = (const float*)d_in[0];   // [8,1024,768]
  const float* qkv_w  = (const float*)d_in[1];   // [768,2304]
  const float* qkv_b  = (const float*)d_in[2];   // [2304]
  const float* proj_w = (const float*)d_in[3];   // [768,768]
  const float* proj_b = (const float*)d_in[4];   // [768]
  float* out = (float*)d_out;                    // [8,1024,768]

  // Workspace layout (halves):
  const size_t QN = (size_t)BATCH * HEADS * SEQ * HDIM;  // 6,291,456
  _Float16* ws = (_Float16*)d_ws;
  _Float16* xH     = ws;            // [8192, 768]
  _Float16* qH     = xH + QN;       // [B*H, N, 64]
  _Float16* kH     = qH + QN;       // [B*H, N, 64]
  _Float16* vtH    = kH + QN;       // [B*H, 64, N]
  _Float16* attnH  = vtH + QN;      // [B, N, 768]
  _Float16* qkvwT  = attnH + QN;    // [2304, 768]  (transposed)
  _Float16* projwT = qkvwT + (size_t)DIM * DQKV;  // [768, 768] (transposed)

  // 0) operand prep: convert x, transpose+convert weights (one-time, L2-sized)
  {
    const int n8 = ROWS * DIM / 8;  // 786,432
    cvt_vec_kernel<<<(n8 + 255) / 256, 256, 0, stream>>>(x, xH, n8);
    cvtT_kernel<<<dim3(DQKV / 32, DIM / 32), 256, 0, stream>>>(qkv_w, qkvwT, DIM, DQKV);
    cvtT_kernel<<<dim3(DIM / 32, DIM / 32), 256, 0, stream>>>(proj_w, projwT, DIM, DIM);
  }
  // 1) QKV projection + scatter to Q/K/V^T
  {
    dim3 grid(DQKV / 128, ROWS / 128);  // (18, 64)
    qkv_gemm_kernel<<<grid, 128, 0, stream>>>(xH, qkvwT, qkv_b, qH, kH, vtH);
  }
  // 2) flash attention
  {
    dim3 grid(SEQ / 64, BATCH * HEADS);  // (16, 96)
    attention_kernel<<<grid, 128, 0, stream>>>(qH, kH, vtH, attnH);
  }
  // 3) output projection (TDM double-buffered LDS pipeline)
  {
    dim3 grid(DIM / 128, ROWS / 128);  // (6, 64)
    proj_gemm_kernel<<<grid, 128, 0, stream>>>(attnH, projwT, proj_b, out);
  }
}
